// OutputModuleX8_67594195305064
// MI455X (gfx1250) — matrix operand order repaired and checked by reading, not compiled, not run on hardware
//
#include <hip/hip_runtime.h>
#include <hip/hip_bf16.h>

// MI455X / gfx1250, wave32. f16 WMMA 16x16x32, fp32 accumulate.
// Async GLOBAL->LDS copies (ASYNCcnt, loads only -- stores share the counter
// but complete out-of-order vs loads, which would break threshold waits).
//
// Pipeline (activations f16, time-major [t][c], c=256):
//   xs  = f16( x[0] + x[1] )                       (256 x 70016, padded)
//   o1  = relu(Wstack1 @ xs + b1)  -> 256 x 140032 (stride-2 deconv as GEMM)
//   o2  = relu(Wstack2 @ o1 + b2)  -> 256 x 280064
//   cls block (16 output times): the reference's mask keeps only k = t%8, so
//   only o3 columns j = 8t+t%8 and j-1 are needed; these derive from just 23
//   o2 columns per block -> stage 3 is computed in-block and o3 never exists.
//   out[c,t] = softmax_c( cW2 @ relu(cW1 @ relu(W4cat @ [o3j; o3jm1] + b4) + cB1) + cB2 )

typedef _Float16 half_t;
typedef __attribute__((ext_vector_type(16))) _Float16 v16h;
typedef __attribute__((ext_vector_type(8)))  _Float16 v8h;
typedef __attribute__((ext_vector_type(8)))  float    v8f;

#define C 256
#define T0 70000
#define TP0 70016            // padded to multiple of 128
#define TP1 (2*TP0)
#define TP2 (4*TP0)

// Low 32 bits of a generic pointer to __shared__ = LDS byte offset
// (ISA: LDS aperture mapping truncates to addr[31:0]).
__device__ __forceinline__ unsigned lds_off(const void* p) {
  return (unsigned)(uintptr_t)p;
}
__device__ __forceinline__ void async_b128(unsigned lds, const void* g) {
  asm volatile("global_load_async_to_lds_b128 %0, %1, off"
               :: "v"(lds), "v"((unsigned long long)(uintptr_t)g) : "memory");
}

// Needed o2 column offsets within a 32-column group: d = 9*(i/3) + 4*(i%3),
// i = 0..10 -> {0,4,8,9,13,17,18,22,26,27,31}. Inverse (d known to be valid):
__device__ __forceinline__ int idx11(int d) {
  return 3 * (d / 9) + (d % 9) / 4;
}
// Slot of o2 column p within a block whose first 8-group is u0.
// Slot 0 = carry-in column 32*u0-1; then 11 slots for group u0, 11 for u0+1.
__device__ __forceinline__ int slot_of(long p, long u0) {
  long u = p >> 5;
  int d = (int)(p & 31);
  int base = (int)(u - (u0 - 1));   // 0,1,2
  return base == 0 ? 0 : 1 + (base - 1) * 11 + idx11(d);
}

// ---------------------------------------------------------------------------
// Weight permutation into A-operand order with contiguous k (this file's choice; B loads match):
//   lane = m_local + 16*(k_local>>4) ; idx = k_local&15
// Tile = 32 lanes x 16 contiguous halfs (1 KB). tileid = mt*KT + kt.
// mode 0: stage dW (C,O,2): A is 512x256, row r = k*256+o, col = c_in. KT=8.
// mode 1: dW4 concat:      A is 256x512, row = o, col = k*256+c_in. KT=16.
// mode 2: cW (O,C):        A is 256x256, row = o, col = c.          KT=8.
__global__ __launch_bounds__(256) void wprep(const float* __restrict__ src,
                                             half_t* __restrict__ dst, int mode) {
  int idx = blockIdx.x * 256 + threadIdx.x;
  int i = idx & 15;
  int l = (idx >> 4) & 31;
  int tileid = idx >> 9;
  int mloc = l & 15;
  int kp = i + 16 * (l >> 4);
  float v = 0.f;
  if (mode == 0) {
    int kt = tileid & 7, mt = tileid >> 3;
    int m = mt * 16 + mloc, c = kt * 32 + kp;
    int k = m >> 8, o = m & 255;
    v = src[c * 512 + o * 2 + k];
  } else if (mode == 1) {
    int kt = tileid & 15, mt = tileid >> 4;
    int o = mt * 16 + mloc, col = kt * 32 + kp;
    int k = col >> 8, c = col & 255;
    v = src[c * 512 + o * 2 + k];
  } else {
    int kt = tileid & 7, mt = tileid >> 3;
    int o = mt * 16 + mloc, c = kt * 32 + kp;
    v = src[o * 256 + c];
  }
  dst[idx] = (half_t)v;
}

// ---------------------------------------------------------------------------
// Sum over S, cast to f16, transpose (c,t) -> time-major (t,c). Zero-pads t>=T0.
__global__ __launch_bounds__(256) void sum_cast(const float* __restrict__ x,
                                                half_t* __restrict__ xs) {
  __shared__ __align__(32) half_t tile[32 * C];
  int tid = threadIdx.x;
  long t0 = (long)blockIdx.x * 32;
  for (int it = 0; it < 32; ++it) {
    int c = it * 8 + (tid >> 5);
    long t = t0 + (tid & 31);
    float v = 0.f;
    if (t < T0)
      v = x[(size_t)c * T0 + t] + x[(size_t)C * T0 + (size_t)c * T0 + t];
    tile[(tid & 31) * C + c] = (half_t)v;
  }
  __syncthreads();
  uint4* dst = (uint4*)(xs + t0 * C);
  const uint4* src = (const uint4*)tile;
  for (int i = 0; i < 4; ++i) dst[tid + 256 * i] = src[tid + 256 * i];
}

// ---------------------------------------------------------------------------
// Stride-2 deconv stage as GEMM: Y(512 x Tin) = Wp(512x256) @ X(256 x Tin),
// rows [0,256) -> output col 2t, rows [256,512) -> col 2t+1. Bias+ReLU fused.
// Block: 256 threads = 8 waves; wave w owns rows [64w,64w+64).
// 4 tiles of 32 time columns, double-buffered via async-to-LDS loads; results
// staged in LDS (ystage == contiguous 64-column span of Y) and drained with
// coalesced b128 stores.
__global__ __launch_bounds__(256) void stage_gemm(const half_t* __restrict__ X,
                                                  const half_t* __restrict__ Wp,
                                                  const float* __restrict__ bias,
                                                  half_t* __restrict__ Y) {
  __shared__ __align__(32) half_t xt[2][32 * C];   // 2 x 16 KB ping-pong
  __shared__ __align__(32) half_t ystage[64 * C];  // 32 KB output staging
  const int tid = threadIdx.x;
  const int wave = tid >> 5, lane = tid & 31;
  const long base = (long)blockIdx.x * 128;
  const int mbase = wave * 64;
  const int kout = wave >> 2;                      // (row >= 256) ? 1 : 0
  const int nlo = lane & 15, khalf = (lane >> 4) * 16, hi = lane >> 4;

  auto tile_async = [&](int buf, long t0) {
    const char* g = (const char*)(X + t0 * C);
    unsigned l0 = lds_off(&xt[buf][0]);
#pragma unroll
    for (int i = 0; i < 4; ++i)
      async_b128(l0 + (unsigned)(tid + 256 * i) * 16, g + (size_t)(tid + 256 * i) * 16);
  };

  tile_async(0, base);
#pragma unroll 1
  for (int it = 0; it < 4; ++it) {
    const int buf = it & 1;
    if (it < 3) {
      tile_async(buf ^ 1, base + (it + 1) * 32);
      // loads-only counter completes in order: <=4 left means current tile landed
      asm volatile("s_wait_asynccnt 0x4" ::: "memory");
    } else {
      asm volatile("s_wait_asynccnt 0x0" ::: "memory");
    }
    __syncthreads();

    v8f acc[4][2] = {};
    for (int kt = 0; kt < 8; ++kt) {
      v16h b0 = *(const v16h*)&xt[buf][nlo * C + kt * 32 + khalf];
      v16h b1 = *(const v16h*)&xt[buf][(nlo + 16) * C + kt * 32 + khalf];
      if (kt < 7)
        __builtin_prefetch(Wp + ((size_t)(mbase >> 4) * 8 + kt + 1) * 512, 0, 3);
#pragma unroll
      for (int mt = 0; mt < 4; ++mt) {
        int tileid = ((mbase >> 4) + mt) * 8 + kt;
        v16h a = *(const v16h*)(Wp + (size_t)tileid * 512 + lane * 16);
        acc[mt][0] = __builtin_amdgcn_wmma_f32_16x16x32_f16(false, a, false, b0,
                                                            (short)0, acc[mt][0], false, false);
        acc[mt][1] = __builtin_amdgcn_wmma_f32_16x16x32_f16(false, a, false, b1,
                                                            (short)0, acc[mt][1], false, false);
      }
    }

    // epilogue: bias + relu, pack 8 consecutive channels -> one b128 LDS store
#pragma unroll
    for (int mt = 0; mt < 4; ++mt)
#pragma unroll
      for (int nt = 0; nt < 2; ++nt) {
        int c0 = (mbase & 255) + mt * 16 + hi * 8;   // 8 consecutive channels
        int outcol = 2 * (nlo + nt * 16) + kout;     // 0..63 within this tile
        v8h h;
#pragma unroll
        for (int r = 0; r < 8; ++r) {
          float v = acc[mt][nt][r] + bias[c0 + r];
          h[r] = (half_t)(v > 0.f ? v : 0.f);
        }
        *(v8h*)&ystage[outcol * C + c0] = h;
      }
    __syncthreads();

    // drain: ystage is exactly Y[2*t0 .. 2*t0+64) x 256, fully coalesced
    const long t0 = base + it * 32;
    uint4* dst = (uint4*)(Y + (size_t)2 * t0 * C);
    const uint4* src = (const uint4*)ystage;
#pragma unroll
    for (int k = 0; k < 8; ++k) dst[tid + 256 * k] = src[tid + 256 * k];
    __syncthreads();  // ystage + xt[buf] reuse protection
  }
}

// ---------------------------------------------------------------------------
// Fused stage3 + deconv4 + classifier + softmax + mask-select.
// Block handles 16 output times t in [t0, t0+16); u0 = t0/8.
// Needs o3 columns j=8t+t%8 and j-1  ->  23 o2 columns (slots, see slot_of).
__global__ __launch_bounds__(256) void cls_kernel(
    const half_t* __restrict__ o2, const half_t* __restrict__ W3p,
    const float* __restrict__ b3, const half_t* __restrict__ W4p,
    const float* __restrict__ b4, const half_t* __restrict__ cW1p,
    const float* __restrict__ cB1, const half_t* __restrict__ cW2p,
    const float* __restrict__ cB2, float* __restrict__ out) {
  __shared__ __align__(32) half_t o2g[32 * C];    // gathered o2 columns (slot-major)
  __shared__ __align__(32) half_t s3[32 * 512];   // stage-3 cols: [slot][k*256+o]
  __shared__ __align__(32) half_t xg[16 * 512];   // [n][k]: o3[j] ; o3[j-1]
  __shared__ __align__(32) half_t g[16 * C];
  __shared__ __align__(32) half_t h1[16 * C];
  __shared__ __align__(32) float  h2[16 * C];
  const int tid = threadIdx.x;
  const int wave = tid >> 5, lane = tid & 31;
  const long t0 = (long)blockIdx.x * 16;
  const long u0 = t0 >> 3;
  const int nlo = lane & 15, khalf = (lane >> 4) * 16, hi = lane >> 4;

  // ---- async-gather the 23 needed o2 columns (512 B each) ----
#pragma unroll
  for (int pass = 0; pass < 3; ++pass) {
    int s = pass * 8 + wave;          // wave-uniform -> EXEC stays full
    if (s < 23) {
      long p;
      if (s == 0) {
        p = 32 * u0 - 1;
        if (p < 0) p = 0;             // t=0: column j-1=-1 is zero-filled below
      } else {
        int gg = (s - 1) / 11, i = (s - 1) % 11;
        p = 32 * (u0 + gg) + 9 * (i / 3) + 4 * (i % 3);
      }
      unsigned d = lds_off(&o2g[s * C]) + (unsigned)lane * 16;
      async_b128(d, (const char*)(o2 + (size_t)p * C) + lane * 16);
    }
  }
  asm volatile("s_wait_asynccnt 0x0" ::: "memory");
  __syncthreads();

  // ---- stage 3 in-block: s3 = relu(Wstack3(512x256) @ o2g + b3), N=32 slots ----
  {
    const int mb = wave * 64;
    v8f acc[4][2] = {};
    for (int kt = 0; kt < 8; ++kt) {
      v16h b0 = *(const v16h*)&o2g[nlo * C + kt * 32 + khalf];
      v16h b1 = *(const v16h*)&o2g[(nlo + 16) * C + kt * 32 + khalf];
#pragma unroll
      for (int mt = 0; mt < 4; ++mt) {
        int tileid = ((mb >> 4) + mt) * 8 + kt;
        v16h a = *(const v16h*)(W3p + (size_t)tileid * 512 + lane * 16);
        acc[mt][0] = __builtin_amdgcn_wmma_f32_16x16x32_f16(false, a, false, b0,
                                                            (short)0, acc[mt][0], false, false);
        acc[mt][1] = __builtin_amdgcn_wmma_f32_16x16x32_f16(false, a, false, b1,
                                                            (short)0, acc[mt][1], false, false);
      }
    }
#pragma unroll
    for (int mt = 0; mt < 4; ++mt)
#pragma unroll
      for (int nt = 0; nt < 2; ++nt) {
        int m0 = mb + mt * 16 + hi * 8;        // 8 consecutive rows
        int slot = nlo + nt * 16;
        v8h h;
#pragma unroll
        for (int r = 0; r < 8; ++r) {
          float v = acc[mt][nt][r] + b3[(m0 + r) & 255];
          h[r] = (half_t)(v > 0.f ? v : 0.f);
        }
        *(v8h*)&s3[slot * 512 + m0] = h;
      }
  }
  __syncthreads();

  // ---- build xg[n][0..511] = [o3[j]; o3[j-1]] from s3 slots ----
  {
    int n = tid >> 4, part = tid & 15;
    long t = t0 + n;
    long j = 8 * t + (t & 7);
    {
      int sj = slot_of(j >> 1, u0);
      const uint4* src = (const uint4*)&s3[sj * 512 + (int)(j & 1) * 256];
      uint4* dst = (uint4*)&xg[n * 512];
      dst[part] = src[part];
      dst[part + 16] = src[part + 16];
    }
    uint4* dst = (uint4*)&xg[n * 512 + 256];
    if (j > 0) {
      long jm = j - 1;
      int sj = slot_of(jm >> 1, u0);
      const uint4* src = (const uint4*)&s3[sj * 512 + (int)(jm & 1) * 256];
      dst[part] = src[part];
      dst[part + 16] = src[part + 16];
    } else {
      uint4 z = {0u, 0u, 0u, 0u};
      dst[part] = z;
      dst[part + 16] = z;
    }
  }
  __syncthreads();

  const int mbase = wave * 32;  // each wave: 2 M-tiles of the 256 output rows

  // GEMM1: g = relu(W4cat(256x512) @ xg + b4)
  {
    v8f acc[2] = {};
    for (int kt = 0; kt < 16; ++kt) {
      v16h b = *(const v16h*)&xg[nlo * 512 + kt * 32 + khalf];
#pragma unroll
      for (int mt = 0; mt < 2; ++mt) {
        int tileid = ((mbase >> 4) + mt) * 16 + kt;
        v16h a = *(const v16h*)(W4p + (size_t)tileid * 512 + lane * 16);
        acc[mt] = __builtin_amdgcn_wmma_f32_16x16x32_f16(false, a, false, b,
                                                         (short)0, acc[mt], false, false);
      }
    }
#pragma unroll
    for (int mt = 0; mt < 2; ++mt)
#pragma unroll
      for (int r = 0; r < 8; ++r) {
        int m = mbase + mt * 16 + r + (hi << 3);
        float v = acc[mt][r] + b4[m];
        g[nlo * C + m] = (half_t)(v > 0.f ? v : 0.f);
      }
  }
  __syncthreads();

  // GEMM2: h1 = relu(cW1 @ g + cB1)
  {
    v8f acc[2] = {};
    for (int kt = 0; kt < 8; ++kt) {
      v16h b = *(const v16h*)&g[nlo * C + kt * 32 + khalf];
#pragma unroll
      for (int mt = 0; mt < 2; ++mt) {
        int tileid = ((mbase >> 4) + mt) * 8 + kt;
        v16h a = *(const v16h*)(cW1p + (size_t)tileid * 512 + lane * 16);
        acc[mt] = __builtin_amdgcn_wmma_f32_16x16x32_f16(false, a, false, b,
                                                         (short)0, acc[mt], false, false);
      }
    }
#pragma unroll
    for (int mt = 0; mt < 2; ++mt)
#pragma unroll
      for (int r = 0; r < 8; ++r) {
        int m = mbase + mt * 16 + r + (hi << 3);
        float v = acc[mt][r] + cB1[m];
        h1[nlo * C + m] = (half_t)(v > 0.f ? v : 0.f);
      }
  }
  __syncthreads();

  // GEMM3: h2 = cW2 @ h1 + cB2  (f32 logits)
  {
    v8f acc[2] = {};
    for (int kt = 0; kt < 8; ++kt) {
      v16h b = *(const v16h*)&h1[nlo * C + kt * 32 + khalf];
#pragma unroll
      for (int mt = 0; mt < 2; ++mt) {
        int tileid = ((mbase >> 4) + mt) * 8 + kt;
        v16h a = *(const v16h*)(cW2p + (size_t)tileid * 512 + lane * 16);
        acc[mt] = __builtin_amdgcn_wmma_f32_16x16x32_f16(false, a, false, b,
                                                         (short)0, acc[mt], false, false);
      }
    }
#pragma unroll
    for (int mt = 0; mt < 2; ++mt)
#pragma unroll
      for (int r = 0; r < 8; ++r) {
        int m = mbase + mt * 16 + r + (hi << 3);
        h2[nlo * C + m] = acc[mt][r] + cB2[m];
      }
  }
  __syncthreads();

  // Softmax over channels per column; 16 lanes cooperate per column.
  {
    int col = tid >> 4;   // time within tile
    int r = tid & 15;
    float vals[16];
    float mx = -3.4e38f;
#pragma unroll
    for (int i = 0; i < 16; ++i) {
      vals[i] = h2[col * C + r + 16 * i];
      mx = fmaxf(mx, vals[i]);
    }
    for (int off = 8; off; off >>= 1) mx = fmaxf(mx, __shfl_xor(mx, off, 16));
    float s = 0.f;
#pragma unroll
    for (int i = 0; i < 16; ++i) {
      vals[i] = __expf(vals[i] - mx);
      s += vals[i];
    }
    for (int off = 8; off; off >>= 1) s += __shfl_xor(s, off, 16);
    float inv = 1.f / s;
    long t = t0 + col;
#pragma unroll
    for (int i = 0; i < 16; ++i)
      out[(size_t)(r + 16 * i) * T0 + t] = vals[i] * inv;
  }
}

// ---------------------------------------------------------------------------
extern "C" void kernel_launch(void* const* d_in, const int* in_sizes, int n_in,
                              void* d_out, int out_size, void* d_ws, size_t ws_size,
                              hipStream_t stream) {
  (void)in_sizes; (void)n_in; (void)out_size; (void)ws_size;
  const float* x   = (const float*)d_in[0];
  const float* dW1 = (const float*)d_in[1];  const float* dB1 = (const float*)d_in[2];
  const float* dW2 = (const float*)d_in[3];  const float* dB2 = (const float*)d_in[4];
  const float* dW3 = (const float*)d_in[5];  const float* dB3 = (const float*)d_in[6];
  const float* dW4 = (const float*)d_in[7];  const float* dB4 = (const float*)d_in[8];
  const float* cW1 = (const float*)d_in[9];  const float* cB1 = (const float*)d_in[10];
  const float* cW2 = (const float*)d_in[11]; const float* cB2 = (const float*)d_in[12];

  char* ws = (char*)d_ws;
  size_t off = 0;
  half_t* xs = (half_t*)(ws + off); off += (size_t)TP0 * C * 2;
  half_t* o1 = (half_t*)(ws + off); off += (size_t)TP1 * C * 2;
  half_t* o2 = (half_t*)(ws + off); off += (size_t)TP2 * C * 2;
  half_t* W1p  = (half_t*)(ws + off); off += 512 * 256 * 2;
  half_t* W2p  = (half_t*)(ws + off); off += 512 * 256 * 2;
  half_t* W3p  = (half_t*)(ws + off); off += 512 * 256 * 2;
  half_t* W4p  = (half_t*)(ws + off); off += 512 * 256 * 2;
  half_t* cW1p = (half_t*)(ws + off); off += 256 * 256 * 2;
  half_t* cW2p = (half_t*)(ws + off); off += 256 * 256 * 2;

  wprep<<<512, 256, 0, stream>>>(dW1, W1p, 0);
  wprep<<<512, 256, 0, stream>>>(dW2, W2p, 0);
  wprep<<<512, 256, 0, stream>>>(dW3, W3p, 0);
  wprep<<<512, 256, 0, stream>>>(dW4, W4p, 1);
  wprep<<<256, 256, 0, stream>>>(cW1, cW1p, 2);
  wprep<<<256, 256, 0, stream>>>(cW2, cW2p, 2);

  sum_cast<<<TP0 / 32, 256, 0, stream>>>(x, xs);

  stage_gemm<<<TP0 / 128, 256, 0, stream>>>(xs, W1p, dB1, o1);
  stage_gemm<<<TP1 / 128, 256, 0, stream>>>(o1, W2p, dB2, o2);

  cls_kernel<<<T0 / 16, 256, 0, stream>>>(o2, W3p, dB3, W4p, dB4,
                                          cW1p, cB1, cW2p, cB2, (float*)d_out);
}
